// UnifiedAttention_77979426226280
// MI455X (gfx1250) — matrix operand order repaired and checked by reading, not compiled, not run on hardware
//
#include <hip/hip_runtime.h>
#include <hip/hip_bf16.h>
#include <stdint.h>

typedef __bf16 bf16_t;
typedef __attribute__((ext_vector_type(16))) __bf16 v16bf;
typedef __attribute__((ext_vector_type(8)))  __bf16 v8bf;
typedef __attribute__((ext_vector_type(8)))  float  v8f;
typedef __attribute__((ext_vector_type(4)))  unsigned int v4u;
typedef __attribute__((ext_vector_type(8)))  int v8i;
typedef __attribute__((ext_vector_type(4)))  int v4i;

#define B_   2
#define F_   128
#define P_   256
#define H_   8
#define D_   64
#define DIM_ 512
#define NQKV 1536   // 3*H*D

#define XSTR    520  // padded stride for X tile in LDS (mult of 8 -> 16B aligned rows)
#define VTSTR_A 136  // padded stride for V^T (frame attention)
#define VTSTR_B 264  // padded stride for V^T (point attention)

__device__ __forceinline__ bf16_t f2bf(float f) {
  union { float f; uint32_t u; } v; v.f = f;
  uint32_t r = v.u + 0x7FFFu + ((v.u >> 16) & 1u);   // round-to-nearest-even
  uint16_t h = (uint16_t)(r >> 16);
  return __builtin_bit_cast(bf16_t, h);
}

__device__ __forceinline__ v8f v8f_zero() {
  v8f z;
#pragma unroll
  for (int j = 0; j < 8; ++j) z[j] = 0.0f;
  return z;
}

// A fragment (16x32 bf16, M x K) from row-major [rows, ld]:
// lane holds row (lane&15); k = k0 + (lane>>4)*8 + {0..7} and +16.
__device__ __forceinline__ v16bf load_frag_a(const bf16_t* __restrict__ base,
                                             int ld, int row0, int k0, int lane) {
  const int r    = lane & 15;
  const int koff = (lane >> 4) << 3;
  const bf16_t* p = base + (size_t)(row0 + r) * ld + k0 + koff;
  v8bf lo = *(const v8bf*)p;
  v8bf hi = *(const v8bf*)(p + 16);
  v16bf out;
#pragma unroll
  for (int i = 0; i < 8; ++i) { out[i] = lo[i]; out[i + 8] = hi[i]; }
  return out;
}

// B fragment (32x16 bf16, K x N) where element (k,n) lives at base[n*ld + k]
// (i.e. an [N,K] row-major matrix): lane holds n = n0 + (lane&15),
// k = k0 + (lane>>4)*8 + {0..7} and +16 (same order as A fragment).
__device__ __forceinline__ v16bf load_frag_b(const bf16_t* __restrict__ base,
                                             int ld, int n0, int k0, int lane) {
  const int n    = lane & 15;
  const int koff = (lane >> 4) << 3;
  const bf16_t* p = base + (size_t)(n0 + n) * ld + k0 + koff;
  v8bf lo = *(const v8bf*)p;
  v8bf hi = *(const v8bf*)(p + 16);
  v16bf out;
#pragma unroll
  for (int i = 0; i < 8; ++i) { out[i] = lo[i]; out[i + 8] = hi[i]; }
  return out;
}

__device__ __forceinline__ v8f wmma_bf16(v16bf a, v16bf b, v8f c) {
  return __builtin_amdgcn_wmma_f32_16x16x32_bf16(false, a, false, b, (short)0, c,
                                                 false, false);
}

// ---------------------------------------------------------------------------
// TDM: DMA a 2D fp32 tile (rows x width, row stride in elements) from global
// memory into LDS. D# layout per CDNA5 ISA ch.8 (group0: count/lds/global
// address/type=2; group1: data_size=4B, tensor dims, tile dims, dim0 stride).
// Tracked with TENSORcnt. Issued once per wave (TDM ignores EXEC).
// ---------------------------------------------------------------------------
__device__ __forceinline__ void tdm_load_2d_f32(uint64_t gaddr, uint32_t lds_byte,
                                                uint32_t width_elems, uint32_t rows,
                                                uint32_t row_stride_elems) {
  v4u g0;
  g0[0] = 1u;                                  // count=1 (valid user descriptor)
  g0[1] = lds_byte;                            // LDS destination (bytes)
  g0[2] = (uint32_t)gaddr;                     // global_addr[31:0]
  g0[3] = ((uint32_t)(gaddr >> 32) & 0x01FFFFFFu) | (2u << 30);  // addr[56:32], type=2
  v8i g1;
  g1[0] = (int)(2u << 16);                     // workgroup_mask=0, data_size=4B
  g1[1] = (int)((width_elems & 0xFFFFu) << 16);                  // tensor_dim0[15:0]
  g1[2] = (int)(((width_elems >> 16) & 0xFFFFu) |
                ((rows & 0xFFFFu) << 16));                       // td0[31:16] | td1[15:0]
  g1[3] = (int)(((rows >> 16) & 0xFFFFu) |
                ((width_elems & 0xFFFFu) << 16));                // td1[31:16] | tile_dim0
  g1[4] = (int)(rows & 0xFFFFu);               // tile_dim1 (tile_dim2 = 0)
  g1[5] = (int)row_stride_elems;               // tensor_dim0_stride[31:0]
  g1[6] = 0;                                   // stride[47:32] | tensor_dim1_stride lo
  g1[7] = 0;
  v4i gz;
#pragma unroll
  for (int i = 0; i < 4; ++i) gz[i] = 0;
#if defined(__clang_major__) && (__clang_major__ >= 23)
  v8i gz8;
#pragma unroll
  for (int i = 0; i < 8; ++i) gz8[i] = 0;
  __builtin_amdgcn_tensor_load_to_lds(g0, g1, gz, gz, gz8, 0);
#else
  __builtin_amdgcn_tensor_load_to_lds(g0, g1, gz, gz, 0);
#endif
}

// ---------------------------------------------------------------------------
// K0: weight conversion. W_f/W_p are [1536,512] (N,K) row-major already.
// W_o is [512(he), 512(d)] -> transpose to [d, he] so it is (N,K) too.
// ---------------------------------------------------------------------------
__global__ void __launch_bounds__(256)
convert_weights_kernel(const float* __restrict__ Wf, const float* __restrict__ Wp,
                       const float* __restrict__ Wo,
                       bf16_t* __restrict__ Wf_b, bf16_t* __restrict__ Wp_b,
                       bf16_t* __restrict__ Wo_t) {
  const int i = blockIdx.x * 256 + threadIdx.x;
  const int NW = NQKV * DIM_;
  if (i < NW) {
    Wf_b[i] = f2bf(Wf[i]);
    Wp_b[i] = f2bf(Wp[i]);
  }
  if (i < DIM_ * DIM_) {
    const int n = i >> 9, k = i & 511;
    Wo_t[(size_t)n * DIM_ + k] = f2bf(Wo[(size_t)k * DIM_ + n]);
  }
}

// ---------------------------------------------------------------------------
// K1: fused frame-QKV projection + frame attention.
// One workgroup per (b, p, h); 4 waves. x tile staged into LDS by the TDM
// (strided 2D gather), converted to bf16, then fully LDS-resident WMMA.
// ---------------------------------------------------------------------------
__global__ void __launch_bounds__(128)
frame_attn_kernel(const float* __restrict__ x, const bf16_t* __restrict__ Wf,
                  const float* __restrict__ bias_f, bf16_t* __restrict__ fa) {
  extern __shared__ bf16_t smem[];
  bf16_t* Xs  = smem;                       // [128][XSTR] bf16 X tile
  // fp32 TDM staging buffer (64 rows x 512 f32 = 128 KB); dead after the X
  // tile is built, so the QKV/P regions below alias it.
  float*  stg = (float*)(Xs + F_ * XSTR);
  bf16_t* Qs  = Xs + F_ * XSTR;             // [128][64]
  bf16_t* Ks  = Qs + F_ * D_;               // [128][64]
  bf16_t* VTs = Ks + F_ * D_;               // [64][VTSTR_A]  (V transposed)
  bf16_t* Ps  = VTs + D_ * VTSTR_A;         // [4 waves][16][128]

  const int wg = blockIdx.x;                // ((b*P + p)*H + h)
  const int h  = wg & (H_ - 1);
  const int bp = wg >> 3;
  const int p  = bp & (P_ - 1);
  const int b  = bp >> 8;
  const int lane = threadIdx.x & 31;
  const int wave = threadIdx.x >> 5;
  const int lr = lane & 15, lh = lane >> 4;

  // ---- stage x[b, :, p, :] -> LDS via TDM, then convert fp32 -> bf16 ----
  const float* xbp = x + ((size_t)b * F_ * P_ + p) * DIM_;
  const uint32_t stg_lds_byte = (uint32_t)(F_ * XSTR * sizeof(bf16_t));
  for (int r = 0; r < 2; ++r) {
    // each wave DMAs its own 16 strided rows (4 descriptors in flight)
    tdm_load_2d_f32(
        (uint64_t)(uintptr_t)(xbp + (size_t)(r * 64 + wave * 16) * P_ * DIM_),
        stg_lds_byte + (uint32_t)wave * 16u * DIM_ * 4u,
        DIM_, 16u, (uint32_t)(P_ * DIM_));
    __builtin_amdgcn_s_wait_tensorcnt(0);
    __syncthreads();
    // cooperative fp32 -> bf16 conversion into the padded X tile
    for (int j = threadIdx.x; j < 64 * (DIM_ / 4); j += 128) {
      const float4 vv = ((const float4*)stg)[j];
      const int row = j >> 7;             // j / (DIM_/4)
      const int col = (j & 127) << 2;
      bf16_t* dst = Xs + (size_t)(r * 64 + row) * XSTR + col;
      dst[0] = f2bf(vv.x); dst[1] = f2bf(vv.y);
      dst[2] = f2bf(vv.z); dst[3] = f2bf(vv.w);
    }
    __syncthreads();
  }

  // ---- QKV projection: each wave owns two 16-row strips ----
  for (int s = 0; s < 2; ++s) {
    const int row0 = (wave * 2 + s) * 16;
    v8f acc[12];
#pragma unroll
    for (int t = 0; t < 12; ++t) acc[t] = v8f_zero();
    for (int kb = 0; kb < 16; ++kb) {
      if (kb < 15)
        __builtin_prefetch((const void*)(Wf + (size_t)(h * D_) * DIM_ + (kb + 1) * 32), 0, 2);
      v16bf af = load_frag_a(Xs, XSTR, row0, kb * 32, lane);
#pragma unroll
      for (int t = 0; t < 12; ++t) {
        const int q = t >> 2, nt = t & 3;
        const int nrow = q * (H_ * D_) + h * D_ + nt * 16;
        v16bf bfr = load_frag_b(Wf, DIM_, nrow, kb * 32, lane);
        acc[t] = wmma_bf16(af, bfr, acc[t]);
      }
    }
#pragma unroll
    for (int t = 0; t < 12; ++t) {
      const int q = t >> 2, nt = t & 3;
      const float bias = bias_f[q * (H_ * D_) + h * D_ + nt * 16 + lr];
      const int col = nt * 16 + lr;
#pragma unroll
      for (int i = 0; i < 8; ++i) {
        const float vv = acc[t][i] + bias;
        const int r = row0 + i + (lh << 3);
        if (q == 0)      Qs[r * D_ + col] = f2bf(vv);
        else if (q == 1) Ks[r * D_ + col] = f2bf(vv);
        else             VTs[col * VTSTR_A + r] = f2bf(vv);   // store V transposed
      }
    }
  }
  __syncthreads();

  // ---- attention across frames (softmax over 128 keys, no 1/sqrt(d)) ----
  for (int s = 0; s < 2; ++s) {
    const int row0 = (wave * 2 + s) * 16;
    v8f sc[8];
#pragma unroll
    for (int t = 0; t < 8; ++t) sc[t] = v8f_zero();
#pragma unroll
    for (int kb = 0; kb < 2; ++kb) {
      v16bf af = load_frag_a(Qs, D_, row0, kb * 32, lane);
#pragma unroll
      for (int nt = 0; nt < 8; ++nt) {
        v16bf bfr = load_frag_b(Ks, D_, nt * 16, kb * 32, lane);
        sc[nt] = wmma_bf16(af, bfr, sc[nt]);
      }
    }
    float mx[8], sm[8];
#pragma unroll
    for (int i = 0; i < 8; ++i) {
      float m = sc[0][i];
#pragma unroll
      for (int nt = 1; nt < 8; ++nt) m = fmaxf(m, sc[nt][i]);
      mx[i] = m;
    }
#pragma unroll
    for (int d = 8; d >= 1; d >>= 1)
#pragma unroll
      for (int i = 0; i < 8; ++i) mx[i] = fmaxf(mx[i], __shfl_xor(mx[i], d, 32));
#pragma unroll
    for (int i = 0; i < 8; ++i) sm[i] = 0.0f;
#pragma unroll
    for (int nt = 0; nt < 8; ++nt)
#pragma unroll
      for (int i = 0; i < 8; ++i) { sc[nt][i] = __expf(sc[nt][i] - mx[i]); sm[i] += sc[nt][i]; }
#pragma unroll
    for (int d = 8; d >= 1; d >>= 1)
#pragma unroll
      for (int i = 0; i < 8; ++i) sm[i] += __shfl_xor(sm[i], d, 32);
#pragma unroll
    for (int i = 0; i < 8; ++i) sm[i] = 1.0f / sm[i];

    bf16_t* Pw = Ps + wave * (16 * 128);
#pragma unroll
    for (int nt = 0; nt < 8; ++nt)
#pragma unroll
      for (int i = 0; i < 8; ++i)
        Pw[(i + (lh << 3)) * 128 + nt * 16 + lr] = f2bf(sc[nt][i] * sm[i]);
    asm volatile("" ::: "memory");   // LDS in-order within wave; stop reordering

    v8f fo[4];
#pragma unroll
    for (int t = 0; t < 4; ++t) fo[t] = v8f_zero();
#pragma unroll
    for (int kb = 0; kb < 4; ++kb) {
      v16bf af = load_frag_a(Pw, 128, 0, kb * 32, lane);
#pragma unroll
      for (int nt = 0; nt < 4; ++nt) {
        v16bf bfr = load_frag_b(VTs, VTSTR_A, nt * 16, kb * 32, lane);
        fo[nt] = wmma_bf16(af, bfr, fo[nt]);
      }
    }
#pragma unroll
    for (int nt = 0; nt < 4; ++nt)
#pragma unroll
      for (int i = 0; i < 8; ++i) {
        const int frame = row0 + i + (lh << 3);
        const size_t off =
            (((size_t)b * F_ + frame) * P_ + p) * DIM_ + h * D_ + nt * 16 + lr;
        fa[off] = f2bf(fo[nt][i]);
      }
    asm volatile("" ::: "memory");
  }
}

// ---------------------------------------------------------------------------
// K2: fused point-QKV projection + point attention.
// One workgroup per (b, f, h); 4 waves; fa A-fragments read from L2.
// ---------------------------------------------------------------------------
__global__ void __launch_bounds__(128)
point_attn_kernel(const bf16_t* __restrict__ fa, const bf16_t* __restrict__ Wp,
                  const float* __restrict__ bias_p, bf16_t* __restrict__ pa) {
  extern __shared__ bf16_t smem[];
  bf16_t* Qs  = smem;                   // [256][64]
  bf16_t* Ks  = Qs + P_ * D_;           // [256][64]
  bf16_t* VTs = Ks + P_ * D_;           // [64][VTSTR_B]
  bf16_t* Ps  = VTs + D_ * VTSTR_B;     // [4 waves][16][256]

  const int wg  = blockIdx.x;           // ((b*F + f)*H + h)
  const int h   = wg & (H_ - 1);
  const int bfi = wg >> 3;
  const int f   = bfi & (F_ - 1);
  const int b   = bfi >> 7;
  const int lane = threadIdx.x & 31;
  const int wave = threadIdx.x >> 5;
  const int lr = lane & 15, lh = lane >> 4;

  const bf16_t* fab = fa + (size_t)(b * F_ + f) * P_ * DIM_;

  // ---- QKV2 projection: each wave owns four 16-row strips of 256 points ----
  for (int s = 0; s < 4; ++s) {
    const int row0 = (wave * 4 + s) * 16;
    v8f acc[12];
#pragma unroll
    for (int t = 0; t < 12; ++t) acc[t] = v8f_zero();
    for (int kb = 0; kb < 16; ++kb) {
      if (kb < 15)
        __builtin_prefetch((const void*)(Wp + (size_t)(h * D_) * DIM_ + (kb + 1) * 32), 0, 2);
      v16bf af = load_frag_a(fab, DIM_, row0, kb * 32, lane);
#pragma unroll
      for (int t = 0; t < 12; ++t) {
        const int q = t >> 2, nt = t & 3;
        const int nrow = q * (H_ * D_) + h * D_ + nt * 16;
        v16bf bfr = load_frag_b(Wp, DIM_, nrow, kb * 32, lane);
        acc[t] = wmma_bf16(af, bfr, acc[t]);
      }
    }
#pragma unroll
    for (int t = 0; t < 12; ++t) {
      const int q = t >> 2, nt = t & 3;
      const float bias = bias_p[q * (H_ * D_) + h * D_ + nt * 16 + lr];
      const int col = nt * 16 + lr;
#pragma unroll
      for (int i = 0; i < 8; ++i) {
        const float vv = acc[t][i] + bias;
        const int r = row0 + i + (lh << 3);
        if (q == 0)      Qs[r * D_ + col] = f2bf(vv);
        else if (q == 1) Ks[r * D_ + col] = f2bf(vv);
        else             VTs[col * VTSTR_B + r] = f2bf(vv);
      }
    }
  }
  __syncthreads();

  // ---- attention across 256 points ----
  for (int s = 0; s < 4; ++s) {
    const int row0 = (wave * 4 + s) * 16;
    v8f sc[16];
#pragma unroll
    for (int t = 0; t < 16; ++t) sc[t] = v8f_zero();
#pragma unroll
    for (int kb = 0; kb < 2; ++kb) {
      v16bf af = load_frag_a(Qs, D_, row0, kb * 32, lane);
#pragma unroll
      for (int nt = 0; nt < 16; ++nt) {
        v16bf bfr = load_frag_b(Ks, D_, nt * 16, kb * 32, lane);
        sc[nt] = wmma_bf16(af, bfr, sc[nt]);
      }
    }
    float mx[8], sm[8];
#pragma unroll
    for (int i = 0; i < 8; ++i) {
      float m = sc[0][i];
#pragma unroll
      for (int nt = 1; nt < 16; ++nt) m = fmaxf(m, sc[nt][i]);
      mx[i] = m;
    }
#pragma unroll
    for (int d = 8; d >= 1; d >>= 1)
#pragma unroll
      for (int i = 0; i < 8; ++i) mx[i] = fmaxf(mx[i], __shfl_xor(mx[i], d, 32));
#pragma unroll
    for (int i = 0; i < 8; ++i) sm[i] = 0.0f;
#pragma unroll
    for (int nt = 0; nt < 16; ++nt)
#pragma unroll
      for (int i = 0; i < 8; ++i) { sc[nt][i] = __expf(sc[nt][i] - mx[i]); sm[i] += sc[nt][i]; }
#pragma unroll
    for (int d = 8; d >= 1; d >>= 1)
#pragma unroll
      for (int i = 0; i < 8; ++i) sm[i] += __shfl_xor(sm[i], d, 32);
#pragma unroll
    for (int i = 0; i < 8; ++i) sm[i] = 1.0f / sm[i];

    bf16_t* Pw = Ps + wave * (16 * P_);
#pragma unroll
    for (int nt = 0; nt < 16; ++nt)
#pragma unroll
      for (int i = 0; i < 8; ++i)
        Pw[(i + (lh << 3)) * P_ + nt * 16 + lr] = f2bf(sc[nt][i] * sm[i]);
    asm volatile("" ::: "memory");

    v8f fo[4];
#pragma unroll
    for (int t = 0; t < 4; ++t) fo[t] = v8f_zero();
#pragma unroll
    for (int kb = 0; kb < 8; ++kb) {
      v16bf af = load_frag_a(Pw, P_, 0, kb * 32, lane);
#pragma unroll
      for (int nt = 0; nt < 4; ++nt) {
        v16bf bfr = load_frag_b(VTs, VTSTR_B, nt * 16, kb * 32, lane);
        fo[nt] = wmma_bf16(af, bfr, fo[nt]);
      }
    }
#pragma unroll
    for (int nt = 0; nt < 4; ++nt)
#pragma unroll
      for (int i = 0; i < 8; ++i) {
        const int point = row0 + i + (lh << 3);
        const size_t off =
            ((size_t)(b * F_ + f) * P_ + point) * DIM_ + h * D_ + nt * 16 + lr;
        pa[off] = f2bf(fo[nt][i]);
      }
    asm volatile("" ::: "memory");
  }
}

// ---------------------------------------------------------------------------
// K3: output projection  out[M=65536, 512] = pa × W_o + b_o  (fp32 out)
// grid (512, 8): 8 waves/WG each own one 16-row strip; blockIdx.y = 64-col slab
// ---------------------------------------------------------------------------
__global__ void __launch_bounds__(256)
out_proj_kernel(const bf16_t* __restrict__ pa, const bf16_t* __restrict__ Wo_t,
                const float* __restrict__ bias_o, float* __restrict__ out) {
  const int lane = threadIdx.x & 31;
  const int wave = threadIdx.x >> 5;
  const int row0 = (blockIdx.x * 8 + wave) * 16;
  const int ncol0 = blockIdx.y * 64;
  const int lr = lane & 15, lh = lane >> 4;

  v8f acc[4];
#pragma unroll
  for (int t = 0; t < 4; ++t) acc[t] = v8f_zero();
  for (int kb = 0; kb < 16; ++kb) {
    v16bf af = load_frag_a(pa, DIM_, row0, kb * 32, lane);
#pragma unroll
    for (int nt = 0; nt < 4; ++nt) {
      v16bf bfr = load_frag_b(Wo_t, DIM_, ncol0 + nt * 16, kb * 32, lane);
      acc[nt] = wmma_bf16(af, bfr, acc[nt]);
    }
  }
#pragma unroll
  for (int nt = 0; nt < 4; ++nt) {
    const int col = ncol0 + nt * 16 + lr;
    const float bias = bias_o[col];
#pragma unroll
    for (int i = 0; i < 8; ++i) {
      const int r = row0 + i + (lh << 3);
      out[(size_t)r * DIM_ + col] = acc[nt][i] + bias;
    }
  }
}

// ---------------------------------------------------------------------------
extern "C" void kernel_launch(void* const* d_in, const int* in_sizes, int n_in,
                              void* d_out, int out_size, void* d_ws, size_t ws_size,
                              hipStream_t stream) {
  const float* x    = (const float*)d_in[0];
  const float* Wf   = (const float*)d_in[1];
  const float* bf_p = (const float*)d_in[2];
  const float* Wp   = (const float*)d_in[3];
  const float* bp_p = (const float*)d_in[4];
  const float* Wo   = (const float*)d_in[5];
  const float* bo_p = (const float*)d_in[6];
  float* out = (float*)d_out;

  // workspace layout (bytes)
  char* ws = (char*)d_ws;
  const size_t szW  = (size_t)NQKV * DIM_ * 2;          // 1.5 MB each
  const size_t szWo = (size_t)DIM_ * DIM_ * 2;          // 0.5 MB
  const size_t szAct = (size_t)B_ * F_ * P_ * DIM_ * 2; // 64 MB each
  bf16_t* Wf_b = (bf16_t*)(ws);
  bf16_t* Wp_b = (bf16_t*)(ws + szW);
  bf16_t* Wo_t = (bf16_t*)(ws + 2 * szW);
  bf16_t* fa   = (bf16_t*)(ws + 2 * szW + szWo);
  bf16_t* pa   = (bf16_t*)(ws + 2 * szW + szWo + szAct);

  // K1 LDS: bf16 X tile + fp32 TDM staging (QKV/P regions alias the staging)
  const size_t smemA = (size_t)(F_ * XSTR + 64 * DIM_ * 2) * sizeof(bf16_t); // ~258 KB
  const size_t smemB = (size_t)(2 * P_ * D_ + D_ * VTSTR_B + 4 * 16 * P_)
                       * sizeof(bf16_t);                                     // ~129 KB
  hipFuncSetAttribute((const void*)frame_attn_kernel,
                      hipFuncAttributeMaxDynamicSharedMemorySize, (int)smemA);
  hipFuncSetAttribute((const void*)point_attn_kernel,
                      hipFuncAttributeMaxDynamicSharedMemorySize, (int)smemB);

  convert_weights_kernel<<<(NQKV * DIM_ + 255) / 256, 256, 0, stream>>>(
      Wf, Wp, Wo, Wf_b, Wp_b, Wo_t);

  frame_attn_kernel<<<B_ * P_ * H_, 128, smemA, stream>>>(x, Wf_b, bf_p, fa);

  point_attn_kernel<<<B_ * F_ * H_, 128, smemB, stream>>>(fa, Wp_b, bp_p, pa);

  out_proj_kernel<<<dim3((B_ * F_ * P_) / (16 * 8), DIM_ / 64), 256, 0, stream>>>(
      pa, Wo_t, bo_p, out);
}
